// Phi3Transformer_54958401519896
// MI455X (gfx1250) — matrix operand order repaired and checked by reading, not compiled, not run on hardware
//
#include <hip/hip_runtime.h>
#include <hip/hip_bf16.h>
#include <math.h>

typedef __attribute__((ext_vector_type(16))) _Float16 v16h;
typedef __attribute__((ext_vector_type(8)))  _Float16 v8h;
typedef __attribute__((ext_vector_type(4)))  _Float16 v4h;
typedef __attribute__((ext_vector_type(8)))  float    v8f;

union U16 { v16h v; v8h h[2]; };

#define HID 2048
#define SEQ 2048
#define NTOK 4096      // B*S
#define LN_EPS 1e-5f

// ---------------------------------------------------------------------------
// WMMA GEMM: C[M,N] = act(alpha * A @ B + bias), batched via blockIdx.z.
//   TRANSB=0: B is [K,N] (row stride ldb)      -> C = A @ B
//   TRANSB=1: B is [N,K] (row stride ldb)      -> C = A @ B^T   (Q·K^T)
// Block = 128 threads (4 waves), 64x64 output tile, BK=32.
// Software-pipelined: global loads of chunk k+1 issue before the WMMAs of
// chunk k; fp32->f16 conversion happens register->LDS with wide DS stores;
// double-buffered LDS gives one barrier per K-chunk.
// ---------------------------------------------------------------------------
template<bool TRANSB, bool RELU>
__global__ __launch_bounds__(128) void gemm_wmma(
    const float* __restrict__ A, const float* __restrict__ B,
    const float* __restrict__ bias, float* __restrict__ C,
    int K, int lda, int ldb, int ldc,
    long sA, long sB, long sC, float alpha)
{
    A += (long)blockIdx.z * sA;
    B += (long)blockIdx.z * sB;
    C += (long)blockIdx.z * sC;

    const int tileM = blockIdx.y * 64;
    const int tileN = blockIdx.x * 64;
    const int tid   = threadIdx.x;
    const int lane  = tid & 31;
    const int wave  = tid >> 5;
    const int wm    = (wave & 1) * 32;   // wave's 32-row quadrant
    const int wn    = (wave >> 1) * 32;  // wave's 32-col quadrant
    const int khalf = lane >> 4;         // lane group (K-half select)
    const int l16   = lane & 15;

    // 40 halfs/row: 80B row stride keeps 16B alignment, skews LDS banks
    __shared__ alignas(16) _Float16 As[2][64][40];   // As[buf][m][k], k in 0..31
    __shared__ alignas(16) _Float16 Bs[2][64][40];   // Bs[buf][n][k], k in 0..31

    v8f acc[2][2] = {};

    // --- staging helpers -------------------------------------------------
    float4 ra[4];        // A chunk: 4 x float4 (row m, 4 consecutive k)
    float4 rb4[4];       // B chunk, TRANSB path (same shape as A)
    float  rb[16];       // B chunk, normal path (column n, 16 consecutive k)
    const int bn  = tid & 63;           // normal-B: column owned by thread
    const int bk0 = (tid >> 6) * 16;    // normal-B: first k owned by thread

    auto loadAB = [&](int kc) {
        #pragma unroll
        for (int it = 0; it < 4; ++it) {
            int idx = tid + it * 128;
            ra[it] = *(const float4*)(A + (long)(tileM + (idx >> 3)) * lda + kc + ((idx & 7) << 2));
        }
        if (TRANSB) {
            #pragma unroll
            for (int it = 0; it < 4; ++it) {
                int idx = tid + it * 128;
                rb4[it] = *(const float4*)(B + (long)(tileN + (idx >> 3)) * ldb + kc + ((idx & 7) << 2));
            }
        } else {
            // lane-coalesced: n varies across lanes, k fixed per load
            #pragma unroll
            for (int u = 0; u < 16; ++u)
                rb[u] = B[(long)(kc + bk0 + u) * ldb + tileN + bn];
        }
    };

    auto storeAB = [&](int b) {
        #pragma unroll
        for (int it = 0; it < 4; ++it) {
            int idx = tid + it * 128;
            int m  = idx >> 3;
            int k4 = (idx & 7) << 2;
            v4h hv;
            hv[0] = (_Float16)ra[it].x; hv[1] = (_Float16)ra[it].y;
            hv[2] = (_Float16)ra[it].z; hv[3] = (_Float16)ra[it].w;
            *(v4h*)&As[b][m][k4] = hv;            // 8-byte DS store
        }
        if (TRANSB) {
            #pragma unroll
            for (int it = 0; it < 4; ++it) {
                int idx = tid + it * 128;
                int n  = idx >> 3;
                int k4 = (idx & 7) << 2;
                v4h hv;
                hv[0] = (_Float16)rb4[it].x; hv[1] = (_Float16)rb4[it].y;
                hv[2] = (_Float16)rb4[it].z; hv[3] = (_Float16)rb4[it].w;
                *(v4h*)&Bs[b][n][k4] = hv;
            }
        } else {
            v8h h0, h1;
            #pragma unroll
            for (int u = 0; u < 8; ++u) { h0[u] = (_Float16)rb[u]; h1[u] = (_Float16)rb[u + 8]; }
            *(v8h*)&Bs[b][bn][bk0]     = h0;      // 16-byte DS stores
            *(v8h*)&Bs[b][bn][bk0 + 8] = h1;
        }
    };

    auto compute = [&](int b) {
        // A 16x32 frag: lane l16 = row; lanes 0-15 hold K {0..7,16..23},
        // lanes 16-31 hold K {8..15,24..31}
        U16 af[2], bf[2];
        #pragma unroll
        for (int i = 0; i < 2; ++i) {
            int m = wm + 16 * i + l16;
            af[i].h[0] = *(const v8h*)&As[b][m][8 * khalf];
            af[i].h[1] = *(const v8h*)&As[b][m][16 + 8 * khalf];
        }
        // B 32x16 frag: lane l16 = col; lanes 0-15 hold K {0..7,16..23}, 16-31 hold {8..15,24..31}
        #pragma unroll
        for (int j = 0; j < 2; ++j) {
            int n = wn + 16 * j + l16;
            bf[j].h[0] = *(const v8h*)&Bs[b][n][8 * khalf];
            bf[j].h[1] = *(const v8h*)&Bs[b][n][16 + 8 * khalf];
        }
        #pragma unroll
        for (int i = 0; i < 2; ++i)
            #pragma unroll
            for (int j = 0; j < 2; ++j)
                acc[i][j] = __builtin_amdgcn_wmma_f32_16x16x32_f16(
                    false, af[i].v, false, bf[j].v, (short)0, acc[i][j], false, false);
    };

    // --- pipelined main loop ---------------------------------------------
    loadAB(0);
    storeAB(0);
    __syncthreads();

    int buf = 0;
    for (int kk = 0; kk < K; kk += 32) {
        const bool more = (kk + 32) < K;
        if (more) loadAB(kk + 32);     // global loads in flight during WMMA
        compute(buf);
        if (more) storeAB(buf ^ 1);    // convert + stage into other buffer
        __syncthreads();
        buf ^= 1;
    }

    // ---- epilogue: C/D layout M = r + 8*(lane/16), N = lane%16 ----
    #pragma unroll
    for (int i = 0; i < 2; ++i) {
        #pragma unroll
        for (int j = 0; j < 2; ++j) {
            int col  = tileN + wn + 16 * j + l16;
            float bv = bias ? bias[col] : 0.0f;
            int row0 = tileM + wm + 16 * i + 8 * khalf;
            #pragma unroll
            for (int r = 0; r < 8; ++r) {
                float val = alpha * acc[i][j][r] + bv;
                if (RELU) val = fmaxf(val, 0.0f);
                C[(long)(row0 + r) * ldc + col] = val;
            }
        }
    }
}

// ---------------------------------------------------------------------------
// Embedding gather: h[row] = wte[ids[row]]
// ---------------------------------------------------------------------------
__global__ __launch_bounds__(256) void embed_kernel(
    const int* __restrict__ ids, const float* __restrict__ wte, float* __restrict__ h)
{
    const int row = blockIdx.x;
    const int id  = ids[row];
    const float4* src = (const float4*)(wte + (long)id * HID);
    float4* dst = (float4*)(h + (long)row * HID);
    for (int i = threadIdx.x; i < HID / 4; i += 256) dst[i] = src[i];
}

// ---------------------------------------------------------------------------
// Row softmax over length SEQ, in place (max-subtracted)
// ---------------------------------------------------------------------------
__global__ __launch_bounds__(256) void softmax_rows(float* __restrict__ scores)
{
    __shared__ float red[256];
    const int tid = threadIdx.x;
    float* row = scores + (long)blockIdx.x * SEQ;

    float m = -INFINITY;
    #pragma unroll
    for (int i = 0; i < SEQ / 256; ++i) m = fmaxf(m, row[tid + i * 256]);
    red[tid] = m; __syncthreads();
    for (int s = 128; s > 0; s >>= 1) {
        if (tid < s) red[tid] = fmaxf(red[tid], red[tid + s]);
        __syncthreads();
    }
    const float mx = red[0]; __syncthreads();

    float sum = 0.0f;
    #pragma unroll
    for (int i = 0; i < SEQ / 256; ++i) {
        int idx = tid + i * 256;
        float e = __expf(row[idx] - mx);
        row[idx] = e;
        sum += e;
    }
    red[tid] = sum; __syncthreads();
    for (int s = 128; s > 0; s >>= 1) {
        if (tid < s) red[tid] += red[tid + s];
        __syncthreads();
    }
    const float inv = 1.0f / red[0];
    #pragma unroll
    for (int i = 0; i < SEQ / 256; ++i) row[tid + i * 256] *= inv;
}

// ---------------------------------------------------------------------------
// out[row] = LayerNorm(x[row] + res[row]) * g + b   (res may be null)
// ---------------------------------------------------------------------------
__global__ __launch_bounds__(256) void ln_res_kernel(
    const float* __restrict__ x, const float* __restrict__ res,
    const float* __restrict__ g, const float* __restrict__ b,
    float* __restrict__ out)
{
    __shared__ float rs[256], rq[256];
    const int tid = threadIdx.x;
    const long base = (long)blockIdx.x * HID;

    float lv[HID / 256];
    float s = 0.0f, sq = 0.0f;
    #pragma unroll
    for (int i = 0; i < HID / 256; ++i) {
        int c = tid + i * 256;
        float v = x[base + c] + (res ? res[base + c] : 0.0f);
        lv[i] = v; s += v; sq += v * v;
    }
    rs[tid] = s; rq[tid] = sq; __syncthreads();
    for (int st = 128; st > 0; st >>= 1) {
        if (tid < st) { rs[tid] += rs[tid + st]; rq[tid] += rq[tid + st]; }
        __syncthreads();
    }
    const float mean = rs[0] / (float)HID;
    const float var  = rq[0] / (float)HID - mean * mean;
    const float rstd = rsqrtf(var + LN_EPS);
    #pragma unroll
    for (int i = 0; i < HID / 256; ++i) {
        int c = tid + i * 256;
        out[base + c] = (lv[i] - mean) * rstd * g[c] + b[c];
    }
}

// ---------------------------------------------------------------------------
extern "C" void kernel_launch(void* const* d_in, const int* in_sizes, int n_in,
                              void* d_out, int out_size, void* d_ws, size_t ws_size,
                              hipStream_t stream) {
    (void)in_sizes; (void)n_in; (void)out_size; (void)ws_size;
    const int*   ids = (const int*)  d_in[0];
    const float* wte = (const float*)d_in[1];
    const float* Wq  = (const float*)d_in[2];
    const float* Wk  = (const float*)d_in[3];
    const float* Wv  = (const float*)d_in[4];
    const float* Wi  = (const float*)d_in[5];
    const float* Wo  = (const float*)d_in[6];
    const float* bq  = (const float*)d_in[7];
    const float* bk  = (const float*)d_in[8];
    const float* bv  = (const float*)d_in[9];
    const float* bi  = (const float*)d_in[10];
    const float* bo  = (const float*)d_in[11];
    const float* g1  = (const float*)d_in[12];
    const float* b1  = (const float*)d_in[13];
    const float* g2  = (const float*)d_in[14];
    const float* b2  = (const float*)d_in[15];
    const float* gf  = (const float*)d_in[16];
    const float* bf  = (const float*)d_in[17];
    float* out = (float*)d_out;

    const long SZ = (long)NTOK * HID;            // one [4096,2048] fp32 buffer
    float* ws     = (float*)d_ws;
    float* h      = ws + 0 * SZ;
    float* q      = ws + 1 * SZ;
    float* kbuf   = ws + 2 * SZ;
    float* vbuf   = ws + 3 * SZ;
    float* scores = ws + 4 * SZ;                 // 2 * 2048 * 2048 == SZ
    float* attn   = ws + 5 * SZ;
    float* h1     = ws + 6 * SZ;
    float* mi     = ws + 7 * SZ;
    float* mo     = ws + 8 * SZ;

    const float scale = 1.0f / sqrtf((float)HID);
    const long  HH    = (long)HID * HID;
    const long  SH    = (long)SEQ * HID;
    const long  SS    = (long)SEQ * SEQ;

    dim3 blk(128);
    dim3 gridW(HID / 64, NTOK / 64, 1);          // dense weight GEMMs, M=4096
    dim3 gridA(SEQ / 64, SEQ / 64, 2);           // batched attention GEMMs

    embed_kernel<<<NTOK, 256, 0, stream>>>(ids, wte, h);

    for (int l = 0; l < 8; ++l) {
        const float* wq = Wq + (long)l * HH;
        const float* wk = Wk + (long)l * HH;
        const float* wv = Wv + (long)l * HH;
        const float* wi = Wi + (long)l * HH;
        const float* wo = Wo + (long)l * HH;

        gemm_wmma<false, false><<<gridW, blk, 0, stream>>>(
            h, wq, bq + l * HID, q, HID, HID, HID, HID, 0, 0, 0, 1.0f);
        gemm_wmma<false, false><<<gridW, blk, 0, stream>>>(
            h, wk, bk + l * HID, kbuf, HID, HID, HID, HID, 0, 0, 0, 1.0f);
        gemm_wmma<false, false><<<gridW, blk, 0, stream>>>(
            h, wv, bv + l * HID, vbuf, HID, HID, HID, HID, 0, 0, 0, 1.0f);

        // scores = (q @ k^T) * 1/sqrt(H), per batch
        gemm_wmma<true, false><<<gridA, blk, 0, stream>>>(
            q, kbuf, nullptr, scores, HID, HID, HID, SEQ, SH, SH, SS, scale);
        softmax_rows<<<2 * SEQ, 256, 0, stream>>>(scores);
        // attn = softmax(scores) @ v, per batch
        gemm_wmma<false, false><<<gridA, blk, 0, stream>>>(
            scores, vbuf, nullptr, attn, SEQ, SEQ, HID, HID, SS, SH, SH, 1.0f);

        ln_res_kernel<<<NTOK, 256, 0, stream>>>(attn, h, g1 + l * HID, b1 + l * HID, h1);

        gemm_wmma<false, true><<<gridW, blk, 0, stream>>>(
            h1, wi, bi + l * HID, mi, HID, HID, HID, HID, 0, 0, 0, 1.0f);
        gemm_wmma<false, false><<<gridW, blk, 0, stream>>>(
            mi, wo, bo + l * HID, mo, HID, HID, HID, HID, 0, 0, 0, 1.0f);

        ln_res_kernel<<<NTOK, 256, 0, stream>>>(mo, h1, g2 + l * HID, b2 + l * HID, h);
    }

    ln_res_kernel<<<NTOK, 256, 0, stream>>>(h, nullptr, gf, bf, out);
}
